// LSTMModule_10926396801446
// MI455X (gfx1250) — compile-verified, matrix operand order checked by reading, not run on hardware
//
#include <hip/hip_runtime.h>
#include <hip/hip_bf16.h>

// ---------------------------------------------------------------------------
// LSTM (B=64, T=512, D=512, H=2048, O=7) for gfx1250 (MI455X, wave32, WMMA)
//
//  - W_ih / W_hh converted to bf16 once (42 MB -> L2-resident across all 512
//    recurrent steps; 192 MB L2 means weights hit HBM ~once).
//  - One fused kernel per timestep. Block = 128 threads = 4 waves; wave g owns
//    gate g for a 16(batch) x 16(hidden) tile -> 2048 waves/step, only 80
//    serial v_wmma_f32_16x16x32_bf16 per wave (short critical path for the
//    latency-bound recurrence).
//  - A = [x_t | h_{t-1}] staged in LDS (16 x 2560 bf16, row-padded 16B for
//    conflict-free 64-bank access), shared by all 4 gate-waves.
//  - Gate results exchanged through 4 KB LDS; sigmoid/tanh + c/h update fused
//    in the same kernel. h stored bf16 in [T+1,B,H] (slot 0 = h0) feeding the
//    next step's A-matrix and the final FC.
// ---------------------------------------------------------------------------

#define B_  64
#define T_  512
#define D_  512
#define H_  2048
#define O_  7
#define G4_ (4 * H_)          // 8192 gate outputs per batch row
#define KX_ (D_ + H_)         // 2560 total K
#define APAD_ 8               // 16B row pad -> conflict-free LDS banks
#define ASTR_ (KX_ + APAD_)   // LDS row stride in bf16 elements

typedef __attribute__((ext_vector_type(16))) __bf16 v16bf;
typedef __attribute__((ext_vector_type(8)))  float  v8f;
typedef __attribute__((ext_vector_type(2)))  __bf16 bf16x2;

union V16 {
    v16bf  v;
    bf16x2 p[8];
    __bf16 e[16];
    uint4  q[2];
};
union V8F {
    v8f   v;
    float e[8];
};

// ---- WMMA fragment loader -------------------------------------------------
// 16-bit A/B 16x32 fragment layout (ISA 7.12.2): lane half (lane>=16) offsets
// K by 8; VGPR v holds K = 2*(v&3) + 16*(v>>2) (+pair). Per lane this is two
// contiguous 16-byte runs at k0+8*half and k0+16+8*half -> two b128 loads
// (ds_load_b128 from LDS, global_load_b128 from memory).
__device__ __forceinline__ v16bf load_frag(const __bf16* rowptr, int k0, int half) {
    V16 u;
    u.q[0] = *reinterpret_cast<const uint4*>(rowptr + k0 + 8 * half);
    u.q[1] = *reinterpret_cast<const uint4*>(rowptr + k0 + 16 + 8 * half);
    return u.v;
}

__device__ __forceinline__ float sigmoid_f(float x) {
    return 1.0f / (1.0f + __expf(-x));
}

// ---- small prep kernels ---------------------------------------------------

__global__ void convert_weights_kernel(const float* __restrict__ Wih,
                                       const float* __restrict__ Whh,
                                       __bf16* __restrict__ Wih_bf,
                                       __bf16* __restrict__ Whh_bf) {
    const size_t n1 = (size_t)G4_ * D_;
    const size_t n2 = (size_t)G4_ * H_;
    const size_t stride = (size_t)gridDim.x * blockDim.x;
    for (size_t i = (size_t)blockIdx.x * blockDim.x + threadIdx.x;
         i < n1 + n2; i += stride) {
        if (i < n1) Wih_bf[i] = (__bf16)Wih[i];
        else        Whh_bf[i - n1] = (__bf16)Whh[i - n1];
    }
}

__global__ void bias_sum_kernel(const float* __restrict__ bih,
                                const float* __restrict__ bhh,
                                float* __restrict__ bias_sum) {
    int i = blockIdx.x * blockDim.x + threadIdx.x;
    if (i < G4_) bias_sum[i] = bih[i] + bhh[i];
}

__global__ void mean_kernel(const float* __restrict__ x,
                            float* __restrict__ mean) {
    int idx = blockIdx.x * blockDim.x + threadIdx.x;   // b*D + d
    if (idx >= B_ * D_) return;
    int b = idx / D_, d = idx % D_;
    float s = 0.f;
    const float* p = x + (size_t)b * T_ * D_ + d;
    for (int t = 0; t < T_; ++t) s += p[(size_t)t * D_];
    mean[idx] = s * (1.0f / T_);
}

__global__ void init_state_kernel(const float* __restrict__ mean,
                                  const float* __restrict__ Wh0,
                                  const float* __restrict__ bh0,
                                  const float* __restrict__ Wc0,
                                  const float* __restrict__ bc0,
                                  __bf16* __restrict__ hs,   // slot 0
                                  float* __restrict__ c) {
    int idx = blockIdx.x * blockDim.x + threadIdx.x;   // b*H + j
    if (idx >= B_ * H_) return;
    int b = idx / H_, j = idx % H_;
    const float* mrow = mean + (size_t)b * D_;
    const float* wh   = Wh0 + (size_t)j * D_;
    const float* wc   = Wc0 + (size_t)j * D_;
    float ah = 0.f, ac = 0.f;
    for (int k = 0; k < D_; ++k) {
        float m = mrow[k];
        ah += m * wh[k];
        ac += m * wc[k];
    }
    hs[idx] = (__bf16)(ah + bh0[j]);
    c[idx]  = ac + bc0[j];
}

// ---- fused per-timestep LSTM kernel ---------------------------------------
// grid = (4 m-tiles, 128 n-tiles); 128 threads = 4 waves, wave g computes
// gate g of the 16x16 (batch x hidden) tile. 2048 waves per step.

__global__ __launch_bounds__(128)
void lstm_step_kernel(const float*  __restrict__ x,        // [B,T,D]
                      const __bf16* __restrict__ Wih_bf,   // [4H,D]
                      const __bf16* __restrict__ Whh_bf,   // [4H,H]
                      const float*  __restrict__ bias_sum, // [4H]
                      float*        __restrict__ c,        // [B,H] fp32
                      __bf16*       __restrict__ hs,       // [(T+1),B,H]
                      int t) {
    __shared__ __bf16 Abuf[16][ASTR_];   // [x_t | h_{t-1}] rows, ~82 KB
    __shared__ float  Cbuf[4][16][16];   // per-gate 16x16 results, 4 KB

    const int tid    = threadIdx.x;
    const int lane   = tid & 31;
    const int wave   = tid >> 5;              // = gate index (i,f,g,o)
    const int half   = lane >> 4;
    const int m_tile = blockIdx.x;            // 0..3  (batch tile)
    const int n_tile = blockIdx.y;            // 0..127 (hidden tile)
    const int m_base = m_tile * 16;
    const int n_col  = n_tile * 16 + (lane & 15);

    // ---- cooperative staging of A into LDS --------------------------------
    // x part: 16 rows x 512 fp32 -> bf16
    for (int idx = tid * 4; idx < 16 * D_; idx += 128 * 4) {
        int row = idx / D_;
        int col = idx % D_;
        const float4 f = *reinterpret_cast<const float4*>(
            x + ((size_t)(m_base + row) * T_ + t) * D_ + col);
        __bf16* dst = &Abuf[row][col];
        dst[0] = (__bf16)f.x; dst[1] = (__bf16)f.y;
        dst[2] = (__bf16)f.z; dst[3] = (__bf16)f.w;
    }
    // h part: 16 rows x 2048 bf16, 16B chunks
    for (int idx = tid * 8; idx < 16 * H_; idx += 128 * 8) {
        int row = idx / H_;
        int col = idx % H_;
        const uint4 q = *reinterpret_cast<const uint4*>(
            hs + ((size_t)t * B_ + (m_base + row)) * H_ + col);
        *reinterpret_cast<uint4*>(&Abuf[row][D_ + col]) = q;
    }
    __syncthreads();

    // ---- GEMM: this wave's gate over full K -------------------------------
    const __bf16* arow = &Abuf[lane & 15][0];
    const __bf16* bih  = Wih_bf + (size_t)(wave * H_ + n_col) * D_;
    const __bf16* bhh  = Whh_bf + (size_t)(wave * H_ + n_col) * H_;

    V8F acc;
    acc.v = (v8f){0.f,0.f,0.f,0.f,0.f,0.f,0.f,0.f};

#pragma unroll 2
    for (int k0 = 0; k0 < D_; k0 += 32) {
        v16bf a = load_frag(arow, k0, half);         // LDS
        v16bf b = load_frag(bih,  k0, half);         // global (L2-resident)
        acc.v = __builtin_amdgcn_wmma_f32_16x16x32_bf16(
            false, a, false, b, (short)0, acc.v, false, false);
    }
#pragma unroll 2
    for (int k0 = 0; k0 < H_; k0 += 32) {
        v16bf a = load_frag(arow + D_, k0, half);    // LDS
        v16bf b = load_frag(bhh,  k0, half);         // global (L2-resident)
        acc.v = __builtin_amdgcn_wmma_f32_16x16x32_bf16(
            false, a, false, b, (short)0, acc.v, false, false);
    }

    // ---- exchange gates through LDS ---------------------------------------
    // C/D layout: element r -> (m = r + 8*half, n = lane&15)
#pragma unroll
    for (int r = 0; r < 8; ++r)
        Cbuf[wave][r + 8 * half][lane & 15] = acc.e[r];
    __syncthreads();

    // ---- fused LSTM update: 256 cells, 2 per thread -----------------------
    for (int i = tid; i < 256; i += 128) {
        int mm = i >> 4;
        int nn = i & 15;
        int b  = m_base + mm;
        int j  = n_tile * 16 + nn;
        size_t idx = (size_t)b * H_ + j;
        float iv = sigmoid_f(Cbuf[0][mm][nn] + bias_sum[j]);
        float fv = sigmoid_f(Cbuf[1][mm][nn] + bias_sum[H_ + j]);
        float gv = tanhf   (Cbuf[2][mm][nn] + bias_sum[2 * H_ + j]);
        float ov = sigmoid_f(Cbuf[3][mm][nn] + bias_sum[3 * H_ + j]);
        float cn = fv * c[idx] + iv * gv;
        c[idx] = cn;
        float hv = ov * tanhf(cn);
        hs[(size_t)(t + 1) * B_ * H_ + idx] = (__bf16)hv;   // slot t+1
    }
}

// ---- final FC: poses[b,t,o] = h_t . W_fc[o] + b_fc[o] ----------------------

__global__ __launch_bounds__(256)
void fc_kernel(const __bf16* __restrict__ hs,   // [(T+1),B,H]
               const float*  __restrict__ Wfc,  // [O,H]
               const float*  __restrict__ bfc,  // [O]
               float*        __restrict__ out)  // [B,T,O]
{
    const int bt = blockIdx.x;        // b*T + t
    const int b  = bt / T_;
    const int t  = bt % T_;
    const __bf16* hrow = hs + ((size_t)(t + 1) * B_ + b) * H_;

    float acc[O_];
#pragma unroll
    for (int o = 0; o < O_; ++o) acc[o] = 0.f;

    for (int j = threadIdx.x; j < H_; j += 256) {
        float hv = (float)hrow[j];
#pragma unroll
        for (int o = 0; o < O_; ++o) acc[o] += hv * Wfc[o * H_ + j];
    }

    __shared__ float red[O_][256];
#pragma unroll
    for (int o = 0; o < O_; ++o) red[o][threadIdx.x] = acc[o];
    __syncthreads();
    for (int s = 128; s > 0; s >>= 1) {
        if ((int)threadIdx.x < s) {
#pragma unroll
            for (int o = 0; o < O_; ++o)
                red[o][threadIdx.x] += red[o][threadIdx.x + s];
        }
        __syncthreads();
    }
    if (threadIdx.x < O_)
        out[(size_t)bt * O_ + threadIdx.x] = red[threadIdx.x][0] + bfc[threadIdx.x];
}

// ---------------------------------------------------------------------------

extern "C" void kernel_launch(void* const* d_in, const int* in_sizes, int n_in,
                              void* d_out, int out_size, void* d_ws, size_t ws_size,
                              hipStream_t stream) {
    (void)in_sizes; (void)n_in; (void)out_size; (void)ws_size;

    const float* x    = (const float*)d_in[0];
    const float* W_ih = (const float*)d_in[1];
    const float* b_ih = (const float*)d_in[2];
    const float* W_hh = (const float*)d_in[3];
    const float* b_hh = (const float*)d_in[4];
    const float* W_h0 = (const float*)d_in[5];
    const float* b_h0 = (const float*)d_in[6];
    const float* W_c0 = (const float*)d_in[7];
    const float* b_c0 = (const float*)d_in[8];
    const float* W_fc = (const float*)d_in[9];
    const float* b_fc = (const float*)d_in[10];
    float* out = (float*)d_out;

    // workspace carve (256B aligned)
    char* ws = (char*)d_ws;
    auto carve = [&](size_t bytes) -> void* {
        void* p = (void*)ws;
        ws += (bytes + 255) & ~(size_t)255;
        return p;
    };
    float*  mean     = (float*) carve((size_t)B_ * D_ * 4);
    float*  c        = (float*) carve((size_t)B_ * H_ * 4);
    float*  bias_sum = (float*) carve((size_t)G4_ * 4);
    __bf16* Wih_bf   = (__bf16*)carve((size_t)G4_ * D_ * 2);
    __bf16* Whh_bf   = (__bf16*)carve((size_t)G4_ * H_ * 2);
    __bf16* hs       = (__bf16*)carve((size_t)(T_ + 1) * B_ * H_ * 2);

    // prep
    convert_weights_kernel<<<4096, 256, 0, stream>>>(W_ih, W_hh, Wih_bf, Whh_bf);
    bias_sum_kernel<<<(G4_ + 255) / 256, 256, 0, stream>>>(b_ih, b_hh, bias_sum);
    mean_kernel<<<(B_ * D_ + 255) / 256, 256, 0, stream>>>(x, mean);
    init_state_kernel<<<(B_ * H_ + 255) / 256, 256, 0, stream>>>(
        mean, W_h0, b_h0, W_c0, b_c0, hs, c);

    // recurrence: one fused WMMA kernel per timestep (graph-captured)
    for (int t = 0; t < T_; ++t) {
        lstm_step_kernel<<<dim3(4, H_ / 16), 128, 0, stream>>>(
            x, Wih_bf, Whh_bf, bias_sum, c, hs, t);
    }

    // output projection (memory-bound: one streaming pass over hs)
    fc_kernel<<<B_ * T_, 256, 0, stream>>>(hs, W_fc, b_fc, out);
}
